// StructureTransformerLayer_54314156425236
// MI455X (gfx1250) — matrix operand order repaired and checked by reading, not compiled, not run on hardware
//
#include <hip/hip_runtime.h>

// ---------------------------------------------------------------------------
// Problem constants (from reference): B=4, L=768, D=512, H=8, DFF=2048, GEOM=7
// ---------------------------------------------------------------------------
#define Bc   4
#define Lc   768
#define Dc   512
#define Hc   8
#define DFFc 2048
#define DKc  64
#define MBr  (Bc*Lc)          // 3072 rows

typedef __attribute__((ext_vector_type(16))) __bf16          v16bf;
typedef __attribute__((ext_vector_type(8)))  float           v8f;
typedef __attribute__((ext_vector_type(16))) unsigned short  u16x16;
typedef __attribute__((ext_vector_type(8)))  unsigned short  u16x8;

__device__ __forceinline__ float bf2f(unsigned short h) {
    unsigned u = ((unsigned)h) << 16;
    return __builtin_bit_cast(float, u);
}
__device__ __forceinline__ unsigned short f2bf(float f) {
    unsigned u = __builtin_bit_cast(unsigned, f);
    u += 0x7FFFu + ((u >> 16) & 1u);           // round-to-nearest-even
    return (unsigned short)(u >> 16);
}
__device__ __forceinline__ float wave_sum(float v) {
    #pragma unroll
    for (int o = 16; o > 0; o >>= 1) v += __shfl_xor(v, o, 32);
    return v;
}
__device__ __forceinline__ float wave_max(float v) {
    #pragma unroll
    for (int o = 16; o > 0; o >>= 1) v = fmaxf(v, __shfl_xor(v, o, 32));
    return v;
}

// ---------------------------------------------------------------------------
// Generic batched bf16 WMMA GEMM, 64x64 register-blocked wave tile:
//   C[b] = A[b](MxK) * Bt[b](NxK)^T (+bias)(+extra)
// Each wave holds 4 A fragments and sweeps 4 B fragments -> 16 WMMAs per
// 32-deep k-step (64 FLOP per cache byte vs 16 for a single-tile wave).
// Fragment layouts used here (contiguous k, not the ISA table order; 16-bit A/B, wave32):
//   A lane l: M = l%16,  K(e) = k0 + e + (l>=16?16:0)
//   B lane l: N = l%16,  K(e) = k0 + e + (l>=16?16:0)
// M, N must be multiples of 64; K a multiple of 32 (true for all uses here).
// ---------------------------------------------------------------------------
__global__ __launch_bounds__(256) void wmma_gemm(
    const unsigned short* __restrict__ A,  long long sA, int lda,
    const unsigned short* __restrict__ Bt, long long sB, int ldb,
    const float* __restrict__ bias,
    float* __restrict__ Cf, unsigned short* __restrict__ Cb,
    long long sC, int ldc,
    const float* __restrict__ extra,
    int M, int N, int K)
{
    int wave = (int)((blockIdx.x * blockDim.x + threadIdx.x) >> 5);
    int lane = threadIdx.x & 31;
    int tN = N >> 6;
    int nTiles = (M >> 6) * tN;
    if (wave >= nTiles) return;                  // wave-uniform branch
    int tm = wave / tN, tn = wave % tN;
    long long batch = blockIdx.y;

    const unsigned short* Ab = A  + batch * sA;
    const unsigned short* Bb = Bt + batch * sB;
    size_t cbase = (size_t)(batch * sC);

    int half = lane >> 4;                        // 0: lanes 0-15, 1: lanes 16-31
    int l16  = lane & 15;

    const unsigned short* ap[4];
    const unsigned short* bp[4];
    #pragma unroll
    for (int i = 0; i < 4; ++i)
        ap[i] = Ab + (size_t)((tm << 6) + (i << 4) + l16) * lda + (half << 4);
    #pragma unroll
    for (int j = 0; j < 4; ++j)
        bp[j] = Bb + (size_t)((tn << 6) + (j << 4) + l16) * ldb + (half << 4);

    v8f acc[4][4] = {};
    for (int k0 = 0; k0 < K; k0 += 32) {
        u16x16 a[4];
        #pragma unroll
        for (int i = 0; i < 4; ++i) {
            u16x8 a0 = *(const u16x8*)(ap[i] + k0);
            u16x8 a1 = *(const u16x8*)(ap[i] + k0 + 8);
            #pragma unroll
            for (int e = 0; e < 8; ++e) { a[i][e] = a0[e]; a[i][e + 8] = a1[e]; }
        }
        #pragma unroll
        for (int j = 0; j < 4; ++j) {
            u16x16 bu = *(const u16x16*)(bp[j] + k0);
            v16bf bv = __builtin_bit_cast(v16bf, bu);
            #pragma unroll
            for (int i = 0; i < 4; ++i) {
                acc[i][j] = __builtin_amdgcn_wmma_f32_16x16x32_bf16(
                    false, __builtin_bit_cast(v16bf, a[i]),
                    false, bv, (short)0, acc[i][j], false, false);
            }
        }
    }

    #pragma unroll
    for (int j = 0; j < 4; ++j) {
        int col = (tn << 6) + (j << 4) + l16;
        float bcol = bias ? bias[col] : 0.0f;
        #pragma unroll
        for (int i = 0; i < 4; ++i) {
            #pragma unroll
            for (int v = 0; v < 8; ++v) {
                int r = (tm << 6) + (i << 4) + v + (half << 3);
                size_t idx = cbase + (size_t)r * ldc + col;
                float val = acc[i][j][v] + bcol;
                if (extra) val += extra[idx];
                if (Cb) Cb[idx] = f2bf(val);
                else    Cf[idx] = val;
            }
        }
    }
}

// ---------------------------------------------------------------------------
// Attention scores: per (b,h) tile of Q·K^T * 1/sqrt(DK) + geomWg bias + mask
// Writes bf16 logits into attnb (B,H,L,L). geom rows are prefetched
// (global_prefetch_b8) before the WMMA loop to overlap the 66 MB stream.
// ---------------------------------------------------------------------------
__global__ __launch_bounds__(256) void scores_kernel(
    const unsigned short* __restrict__ Qb, const unsigned short* __restrict__ Kb,
    const float* __restrict__ geom, const float* __restrict__ Wg,
    const float* __restrict__ bgv, const int* __restrict__ mask,
    unsigned short* __restrict__ attnb)
{
    int wave = (int)((blockIdx.x * blockDim.x + threadIdx.x) >> 5);
    int lane = threadIdx.x & 31;
    const int tL = Lc / 16;
    if (wave >= tL * tL) return;
    int bh = blockIdx.y;
    int b = bh >> 3, h = bh & 7;
    int tm = wave / tL, tn = wave % tL;
    int half = lane >> 4, l16 = lane & 15;
    int row = tm * 16 + l16;                     // l
    int col = tn * 16 + l16;                     // m

    // prefetch the geom rows this lane's epilogue will read
    #pragma unroll
    for (int v = 0; v < 8; ++v) {
        int l = tm * 16 + v + (half << 3);
        __builtin_prefetch(geom + (((size_t)(b * Lc + l)) * Lc + col) * 7, 0, 1);
    }

    const unsigned short* ap = Qb + ((size_t)(b * Lc + row)) * Dc + h * DKc + (half << 4);
    const unsigned short* bp = Kb + ((size_t)(b * Lc + col)) * Dc + h * DKc + (half << 4);

    v8f acc = {};
    #pragma unroll
    for (int k0 = 0; k0 < DKc; k0 += 32) {
        u16x8  a0 = *(const u16x8*)(ap + k0);
        u16x8  a1 = *(const u16x8*)(ap + k0 + 8);
        u16x16 bu = *(const u16x16*)(bp + k0);
        u16x16 au;
        #pragma unroll
        for (int i = 0; i < 8; ++i) { au[i] = a0[i]; au[i + 8] = a1[i]; }
        acc = __builtin_amdgcn_wmma_f32_16x16x32_bf16(
                  false, __builtin_bit_cast(v16bf, au),
                  false, __builtin_bit_cast(v16bf, bu),
                  (short)0, acc, false, false);
    }

    float wg[7];
    #pragma unroll
    for (int g = 0; g < 7; ++g) wg[g] = Wg[g * Hc + h];   // h wave-uniform -> s_loads
    float bgh = bgv[h];
    int mcol = mask[b * Lc + col];
    const float scale = 0.125f;                            // 1/sqrt(64)

    #pragma unroll
    for (int v = 0; v < 8; ++v) {
        int l = tm * 16 + v + (half << 3);
        const float* gp = geom + (((size_t)(b * Lc + l)) * Lc + col) * 7;
        float bias_ = bgh;
        #pragma unroll
        for (int g = 0; g < 7; ++g) bias_ += gp[g] * wg[g];
        float s = acc[v] * scale + bias_;
        if (mcol == 0) s = -10000.0f;
        attnb[((size_t)bh * Lc + l) * Lc + col] = f2bf(s);
    }
}

// One wave per row of 768 logits: in-place softmax (bf16 storage, f32 math)
__global__ __launch_bounds__(256) void softmax_kernel(unsigned short* __restrict__ attnb)
{
    int row  = (int)((blockIdx.x * blockDim.x + threadIdx.x) >> 5);
    int lane = threadIdx.x & 31;
    if (row >= Bc * Hc * Lc) return;
    unsigned short* p = attnb + (size_t)row * Lc;
    float vals[24]; float mx = -3.4e38f;
    #pragma unroll
    for (int i = 0; i < 24; ++i) { vals[i] = bf2f(p[lane + i * 32]); mx = fmaxf(mx, vals[i]); }
    mx = wave_max(mx);
    float sum = 0.f;
    #pragma unroll
    for (int i = 0; i < 24; ++i) { vals[i] = __expf(vals[i] - mx); sum += vals[i]; }
    sum = wave_sum(sum);
    float inv = 1.0f / sum;
    #pragma unroll
    for (int i = 0; i < 24; ++i) p[lane + i * 32] = f2bf(vals[i] * inv);
}

// LayerNorm 1: one wave per 512-wide row, x(f32) -> hb(bf16)
__global__ __launch_bounds__(256) void ln1_kernel(
    const float* __restrict__ x, const float* __restrict__ sc,
    const float* __restrict__ bi, unsigned short* __restrict__ hb)
{
    int row  = (int)((blockIdx.x * blockDim.x + threadIdx.x) >> 5);
    int lane = threadIdx.x & 31;
    if (row >= MBr) return;
    const float* xp = x + (size_t)row * Dc;
    float v[16]; float s = 0.f;
    #pragma unroll
    for (int i = 0; i < 16; ++i) { v[i] = xp[lane + i * 32]; s += v[i]; }
    float mu = wave_sum(s) * (1.0f / Dc);
    float var = 0.f;
    #pragma unroll
    for (int i = 0; i < 16; ++i) { float d = v[i] - mu; var += d * d; }
    var = wave_sum(var) * (1.0f / Dc);
    float rs = rsqrtf(var + 1e-5f);
    #pragma unroll
    for (int i = 0; i < 16; ++i) {
        int c = lane + i * 32;
        hb[(size_t)row * Dc + c] = f2bf((v[i] - mu) * rs * sc[c] + bi[c]);
    }
}

// Residual add (x + attn_out + gate*gelu(gcnlin)) -> xmid(f32), LN2 -> h2b(bf16)
__global__ __launch_bounds__(256) void ln2_kernel(
    const float* __restrict__ x, const float* __restrict__ ao,
    const float* __restrict__ gl, const float* __restrict__ gatep,
    const float* __restrict__ sc, const float* __restrict__ bi,
    float* __restrict__ xmid, unsigned short* __restrict__ h2b)
{
    int row  = (int)((blockIdx.x * blockDim.x + threadIdx.x) >> 5);
    int lane = threadIdx.x & 31;
    if (row >= MBr) return;
    float gate = gatep[0];
    float v[16]; float s = 0.f;
    #pragma unroll
    for (int i = 0; i < 16; ++i) {
        int c = lane + i * 32;
        size_t idx = (size_t)row * Dc + c;
        float t  = gl[idx];
        float ge = 0.5f * t * (1.0f + erff(t * 0.70710678f));   // exact gelu
        float xv = x[idx] + ao[idx] + gate * ge;
        xmid[idx] = xv; v[i] = xv; s += xv;
    }
    float mu = wave_sum(s) * (1.0f / Dc);
    float var = 0.f;
    #pragma unroll
    for (int i = 0; i < 16; ++i) { float d = v[i] - mu; var += d * d; }
    var = wave_sum(var) * (1.0f / Dc);
    float rs = rsqrtf(var + 1e-5f);
    #pragma unroll
    for (int i = 0; i < 16; ++i) {
        int c = lane + i * 32;
        h2b[(size_t)row * Dc + c] = f2bf((v[i] - mu) * rs * sc[c] + bi[c]);
    }
}

// adj row: exp(-geom0/5)*m_l*m_m, row-normalized -> bf16 (B,L,L)
__global__ __launch_bounds__(256) void adj_kernel(
    const float* __restrict__ geom, const int* __restrict__ mask,
    unsigned short* __restrict__ adjb)
{
    int row  = (int)((blockIdx.x * blockDim.x + threadIdx.x) >> 5);   // b*L + l
    int lane = threadIdx.x & 31;
    if (row >= Bc * Lc) return;
    int b = row / Lc;
    float ml = (float)mask[row];
    float vals[24]; float s = 0.f;
    #pragma unroll
    for (int i = 0; i < 24; ++i) {
        int m = lane + i * 32;
        float g0 = geom[(((size_t)row) * Lc + m) * 7];
        float w  = __expf(-0.2f * g0) * ml * (float)mask[b * Lc + m];
        vals[i] = w; s += w;
    }
    s = wave_sum(s);
    float inv = 1.0f / (s + 1e-6f);
    #pragma unroll
    for (int i = 0; i < 24; ++i)
        adjb[((size_t)row) * Lc + lane + i * 32] = f2bf(vals[i] * inv);
}

// bf16 (B,L,D) -> (B,D,L) transpose (used for V and g)
__global__ __launch_bounds__(256) void transpose_kernel(
    const unsigned short* __restrict__ in, unsigned short* __restrict__ out)
{
    size_t i = (size_t)blockIdx.x * 256 + threadIdx.x;
    if (i >= (size_t)Bc * Lc * Dc) return;
    int d = (int)(i % Dc); size_t bl = i / Dc;
    int l = (int)(bl % Lc); int b = (int)(bl / Lc);
    out[((size_t)b * Dc + d) * Lc + l] = in[i];
}

// ctx (B,H,L,DK) -> ctxp (B,L,D)
__global__ __launch_bounds__(256) void ctx_repack_kernel(
    const unsigned short* __restrict__ ctx, unsigned short* __restrict__ ctxp)
{
    size_t i = (size_t)blockIdx.x * 256 + threadIdx.x;
    if (i >= (size_t)Bc * Hc * Lc * DKc) return;
    int dk = (int)(i & 63); size_t t = i >> 6;
    int l = (int)(t % Lc); t /= Lc;
    int h = (int)(t & 7);  int b = (int)(t >> 3);
    ctxp[((size_t)(b * Lc + l)) * Dc + h * DKc + dk] = ctx[i];
}

// Weight convert: W (KxN f32 row-major) -> Wt (NxK bf16)
__global__ __launch_bounds__(256) void wcvt_kernel(
    const float* __restrict__ W, unsigned short* __restrict__ Wt, int K, int N)
{
    size_t i = (size_t)blockIdx.x * 256 + threadIdx.x;
    if (i >= (size_t)K * N) return;
    int n = (int)(i % N); int k = (int)(i / N);
    Wt[(size_t)n * K + k] = f2bf(W[i]);
}

// GLU activation: u * sigmoid(gate) from ff (3072 x 4096) -> actb (3072 x 2048)
__global__ __launch_bounds__(256) void act_kernel(
    const unsigned short* __restrict__ ffb, unsigned short* __restrict__ actb)
{
    size_t i = (size_t)blockIdx.x * 256 + threadIdx.x;
    if (i >= (size_t)MBr * DFFc) return;
    int j = (int)(i % DFFc); size_t r = i / DFFc;
    float u = bf2f(ffb[r * (2 * DFFc) + j]);
    float g = bf2f(ffb[r * (2 * DFFc) + DFFc + j]);
    actb[i] = f2bf(u / (1.0f + __expf(-g)));
}

// ---------------------------------------------------------------------------
extern "C" void kernel_launch(void* const* d_in, const int* in_sizes, int n_in,
                              void* d_out, int out_size, void* d_ws, size_t ws_size,
                              hipStream_t stream)
{
    (void)in_sizes; (void)n_in; (void)out_size; (void)ws_size;
    const float* x    = (const float*)d_in[0];
    const float* geom = (const float*)d_in[1];
    const int*   mask = (const int*)  d_in[2];
    const float* Wq   = (const float*)d_in[3];  const float* bq   = (const float*)d_in[4];
    const float* Wk   = (const float*)d_in[5];  const float* bk   = (const float*)d_in[6];
    const float* Wv   = (const float*)d_in[7];  const float* bv   = (const float*)d_in[8];
    const float* Wo   = (const float*)d_in[9];  const float* bo   = (const float*)d_in[10];
    const float* Wg   = (const float*)d_in[11]; const float* bg   = (const float*)d_in[12];
    const float* Wgcn = (const float*)d_in[13]; const float* bgcn = (const float*)d_in[14];
    const float* gate = (const float*)d_in[15];
    const float* W1   = (const float*)d_in[16]; const float* b1   = (const float*)d_in[17];
    const float* W2   = (const float*)d_in[18]; const float* b2   = (const float*)d_in[19];
    const float* ln1s = (const float*)d_in[20]; const float* ln1b = (const float*)d_in[21];
    const float* ln2s = (const float*)d_in[22]; const float* ln2b = (const float*)d_in[23];
    float* outp = (float*)d_out;

    char* ws = (char*)d_ws;
    size_t off = 0;
    auto alloc = [&](size_t bytes) { size_t p = off; off += (bytes + 255) & ~(size_t)255; return p; };
    auto US = [&](size_t o) { return (unsigned short*)(ws + o); };
    auto FP = [&](size_t o) { return (float*)(ws + o); };

    // persistent
    size_t oWqT   = alloc((size_t)Dc * Dc * 2);
    size_t oWkT   = alloc((size_t)Dc * Dc * 2);
    size_t oWvT   = alloc((size_t)Dc * Dc * 2);
    size_t oWoT   = alloc((size_t)Dc * Dc * 2);
    size_t oWgcnT = alloc((size_t)Dc * Dc * 2);
    size_t oW1T   = alloc((size_t)Dc * 2 * DFFc * 2);
    size_t oW2T   = alloc((size_t)DFFc * Dc * 2);
    size_t oXmid  = alloc((size_t)MBr * Dc * 4);
    size_t stage  = off;
    // stage A (attention + gcn)
    size_t oHb    = alloc((size_t)MBr * Dc * 2);
    size_t oQb    = alloc((size_t)MBr * Dc * 2);
    size_t oKb    = alloc((size_t)MBr * Dc * 2);
    size_t oVb    = alloc((size_t)MBr * Dc * 2);
    size_t oVt    = alloc((size_t)Bc * Dc * Lc * 2);
    size_t oAttn  = alloc((size_t)Bc * Hc * Lc * Lc * 2);
    size_t oCtx   = alloc((size_t)Bc * Hc * Lc * DKc * 2);
    size_t oCtxp  = alloc((size_t)MBr * Dc * 2);
    size_t oAo    = alloc((size_t)MBr * Dc * 4);
    size_t oAdj   = alloc((size_t)Bc * Lc * Lc * 2);
    size_t oGb    = alloc((size_t)MBr * Dc * 2);
    size_t oGt    = alloc((size_t)Bc * Dc * Lc * 2);
    size_t oGcn   = alloc((size_t)MBr * Dc * 4);
    // stage B (ffn) overlays stage A (all stage-A tensors dead by then)
    off = stage;
    size_t oH2b   = alloc((size_t)MBr * Dc * 2);
    size_t oFfb   = alloc((size_t)MBr * 2 * DFFc * 2);
    size_t oActb  = alloc((size_t)MBr * DFFc * 2);

    dim3 blk(256);
    auto gblocks = [](size_t n) { return dim3((unsigned)((n + 255) / 256)); };
    auto gemmgrid = [](int M, int N, int batches) {
        int t = (M / 64) * (N / 64);
        return dim3((unsigned)((t + 7) / 8), (unsigned)batches);
    };

    // 1) weight transpose + bf16 convert
    wcvt_kernel<<<gblocks((size_t)Dc * Dc), blk, 0, stream>>>(Wq,   US(oWqT),   Dc, Dc);
    wcvt_kernel<<<gblocks((size_t)Dc * Dc), blk, 0, stream>>>(Wk,   US(oWkT),   Dc, Dc);
    wcvt_kernel<<<gblocks((size_t)Dc * Dc), blk, 0, stream>>>(Wv,   US(oWvT),   Dc, Dc);
    wcvt_kernel<<<gblocks((size_t)Dc * Dc), blk, 0, stream>>>(Wo,   US(oWoT),   Dc, Dc);
    wcvt_kernel<<<gblocks((size_t)Dc * Dc), blk, 0, stream>>>(Wgcn, US(oWgcnT), Dc, Dc);
    wcvt_kernel<<<gblocks((size_t)Dc * 2 * DFFc), blk, 0, stream>>>(W1, US(oW1T), Dc, 2 * DFFc);
    wcvt_kernel<<<gblocks((size_t)DFFc * Dc), blk, 0, stream>>>(W2, US(oW2T), DFFc, Dc);

    // 2) LN1
    ln1_kernel<<<dim3(MBr / 8), blk, 0, stream>>>(x, ln1s, ln1b, US(oHb));

    // 3) Q,K,V projections (M=3072,N=512,K=512)
    wmma_gemm<<<gemmgrid(MBr, Dc, 1), blk, 0, stream>>>(US(oHb), 0, Dc, US(oWqT), 0, Dc, bq,
        nullptr, US(oQb), 0, Dc, nullptr, MBr, Dc, Dc);
    wmma_gemm<<<gemmgrid(MBr, Dc, 1), blk, 0, stream>>>(US(oHb), 0, Dc, US(oWkT), 0, Dc, bk,
        nullptr, US(oKb), 0, Dc, nullptr, MBr, Dc, Dc);
    wmma_gemm<<<gemmgrid(MBr, Dc, 1), blk, 0, stream>>>(US(oHb), 0, Dc, US(oWvT), 0, Dc, bv,
        nullptr, US(oVb), 0, Dc, nullptr, MBr, Dc, Dc);

    // 4) V -> Vt (B,D,L)
    transpose_kernel<<<gblocks((size_t)Bc * Lc * Dc), blk, 0, stream>>>(US(oVb), US(oVt));

    // 5) scores + geom bias + mask (per b,h); 2304 16x16 tiles -> 288 blocks, 32 batches
    scores_kernel<<<dim3(288, Bc * Hc), blk, 0, stream>>>(US(oQb), US(oKb), geom, Wg, bg, mask, US(oAttn));

    // 6) softmax over 24576 rows
    softmax_kernel<<<dim3(Bc * Hc * Lc / 8), blk, 0, stream>>>(US(oAttn));

    // 7) ctx = attn @ V per (b,h): M=768,N=64,K=768
    wmma_gemm<<<gemmgrid(Lc, DKc, Bc * Hc), blk, 0, stream>>>(
        US(oAttn), (long long)Lc * Lc, Lc,
        US(oVt),   (long long)DKc * Lc, Lc,
        nullptr, nullptr, US(oCtx), (long long)Lc * DKc, DKc, nullptr, Lc, DKc, Lc);

    // 8) ctx -> (B,L,D)
    ctx_repack_kernel<<<gblocks((size_t)Bc * Hc * Lc * DKc), blk, 0, stream>>>(US(oCtx), US(oCtxp));

    // 9) attn_out = ctxp @ Wo + bo (f32)
    wmma_gemm<<<gemmgrid(MBr, Dc, 1), blk, 0, stream>>>(US(oCtxp), 0, Dc, US(oWoT), 0, Dc, bo,
        FP(oAo), nullptr, 0, Dc, nullptr, MBr, Dc, Dc);

    // 10) g = h @ Wgcn + bgcn (bf16), then transpose
    wmma_gemm<<<gemmgrid(MBr, Dc, 1), blk, 0, stream>>>(US(oHb), 0, Dc, US(oWgcnT), 0, Dc, bgcn,
        nullptr, US(oGb), 0, Dc, nullptr, MBr, Dc, Dc);
    transpose_kernel<<<gblocks((size_t)Bc * Lc * Dc), blk, 0, stream>>>(US(oGb), US(oGt));

    // 11) adj rows
    adj_kernel<<<dim3(Bc * Lc / 8), blk, 0, stream>>>(geom, mask, US(oAdj));

    // 12) gcnlin = adj @ g per batch: M=768,N=512,K=768
    wmma_gemm<<<gemmgrid(Lc, Dc, Bc), blk, 0, stream>>>(
        US(oAdj), (long long)Lc * Lc, Lc,
        US(oGt),  (long long)Dc * Lc, Lc,
        nullptr, FP(oGcn), nullptr, (long long)Lc * Dc, Dc, nullptr, Lc, Dc, Lc);

    // 13) residual + gelu + LN2
    ln2_kernel<<<dim3(MBr / 8), blk, 0, stream>>>(x, FP(oAo), FP(oGcn), gate, ln2s, ln2b,
                                                  FP(oXmid), US(oH2b));

    // 14) ff = h2 @ W1 + b1 (M=3072,N=4096,K=512)
    wmma_gemm<<<gemmgrid(MBr, 2 * DFFc, 1), blk, 0, stream>>>(US(oH2b), 0, Dc, US(oW1T), 0, Dc, b1,
        nullptr, US(oFfb), 0, 2 * DFFc, nullptr, MBr, 2 * DFFc, Dc);

    // 15) GLU activation
    act_kernel<<<gblocks((size_t)MBr * DFFc), blk, 0, stream>>>(US(oFfb), US(oActb));

    // 16) out = xmid + act @ W2 + b2 (M=3072,N=512,K=2048)
    wmma_gemm<<<gemmgrid(MBr, Dc, 1), blk, 0, stream>>>(US(oActb), 0, DFFc, US(oW2T), 0, DFFc, b2,
        outp, nullptr, 0, Dc, FP(oXmid), MBr, Dc, DFFc);
}
